// InnerAttention_78417512891098
// MI455X (gfx1250) — matrix operand order repaired and checked by reading, not compiled, not run on hardware
//
#include <hip/hip_runtime.h>

#define B_   4
#define S_   2048
#define D_   1024
#define H_   16
#define HD_  64
#define WIN_ 256

typedef __attribute__((ext_vector_type(16))) __bf16 v16bf;
typedef __attribute__((ext_vector_type(8)))  float  v8f;
typedef __attribute__((ext_vector_type(2)))  __bf16 bf16x2;
typedef __attribute__((ext_vector_type(4)))  __bf16 bf16x4;
typedef __attribute__((ext_vector_type(4)))  float  f32x4;

static __device__ __forceinline__ v8f wmma_bf16(v16bf a, v16bf b, v8f c) {
  // D = A(16x32 bf16) x B(32x16 bf16) + C(16x16 f32)
  return __builtin_amdgcn_wmma_f32_16x16x32_bf16(false, a, false, b, (short)0, c,
                                                 false, false);
}

static __device__ __forceinline__ unsigned lds_off32(const void* p) {
  // gfx1250 aperture rule: for LDS generic pointers, the LDS address is the low 32 bits of the pointer
  return (unsigned)(uintptr_t)p;
}

// ---------------------------------------------------------------------------
// Elementwise f32 -> bf16 conversion (n multiple of 4)
// ---------------------------------------------------------------------------
__global__ __launch_bounds__(256) void cvt_f32_bf16(const float* __restrict__ src,
                                                    __bf16* __restrict__ dst,
                                                    int n) {
  const int i = (blockIdx.x * 256 + threadIdx.x) * 4;
  if (i < n) {
    f32x4 v = *(const f32x4*)(src + i);
    bf16x4 o;
    o[0] = (__bf16)v[0]; o[1] = (__bf16)v[1];
    o[2] = (__bf16)v[2]; o[3] = (__bf16)v[3];
    *(bf16x4*)(dst + i) = o;
  }
}

// ---------------------------------------------------------------------------
// GEMM: out[M,N] = A[M,K] @ W[N,K]^T, all-bf16 inputs, f32 WMMA accum.
// Block tile 128x128, 256 threads (8 waves), wave computes 64x32.
// Double-buffered LDS: A tile staged with async global->LDS (ASYNCcnt),
// W tile prefetched to registers and transposed into LDS. 1 barrier/K-step.
// ---------------------------------------------------------------------------
template <bool OUT_BF16>
__global__ __launch_bounds__(256) void gemm_xwT(const __bf16* __restrict__ A,
                                                const __bf16* __restrict__ W,
                                                void* __restrict__ out,
                                                const float* __restrict__ bias,
                                                int M, int N, int K) {
  __shared__ __align__(16) __bf16 As[2][128 * 40];  // [m][k], stride 40 (80B)
  __shared__ __align__(16) __bf16 Bs[2][32 * 130];  // [k][n], stride 130

  const int tid  = threadIdx.x;
  const int lane = tid & 31;
  const int wave = tid >> 5;
  const int m0 = blockIdx.x * 128;
  const int n0 = blockIdx.y * 128;
  const int wr = wave >> 2;
  const int wc = wave & 3;

  v8f acc[4][2];
  for (int i = 0; i < 4; i++)
    for (int j = 0; j < 2; j++)
      for (int e = 0; e < 8; e++) acc[i][j][e] = 0.0f;

  const int r  = tid >> 3;       // 0..31  (W staging row group)
  const int kc = (tid & 7) * 4;  // 0..28  (W staging k offset)
  const int mhalf = lane >> 4;
  const int ml = lane & 15;

  // Async A-tile staging: 256 threads x 32B; row = tid/2, 16-col half = tid&1
  const int arow  = tid >> 1;
  const int ahalf = (tid & 1) * 16;

#define STAGE_A(buf, kk0)                                                      \
  do {                                                                         \
    const __bf16* g_ = A + (size_t)(m0 + arow) * K + (kk0) + ahalf;            \
    unsigned l_ = lds_off32(&As[buf][arow * 40 + ahalf]);                      \
    asm volatile("global_load_async_to_lds_b128 %0, %1, off\n\t"               \
                 "global_load_async_to_lds_b128 %0, %1, off offset:16"         \
                 :: "v"(l_), "v"(g_) : "memory");                              \
  } while (0)

#define WAIT_ASYNC() asm volatile("s_wait_asynccnt 0x0" ::: "memory")

  bf16x4 wreg[4];
#define LOAD_W(kk0)                                                            \
  do {                                                                         \
    _Pragma("unroll") for (int rr = 0; rr < 4; rr++) {                         \
      const int row_ = rr * 32 + r;                                            \
      wreg[rr] = *(const bf16x4*)(W + (size_t)(n0 + row_) * K + (kk0) + kc);   \
    }                                                                          \
  } while (0)
#define STORE_W(buf)                                                           \
  do {                                                                         \
    _Pragma("unroll") for (int rr = 0; rr < 4; rr++) {                         \
      const int row_ = rr * 32 + r;                                            \
      Bs[buf][(kc + 0) * 130 + row_] = wreg[rr][0];                            \
      Bs[buf][(kc + 1) * 130 + row_] = wreg[rr][1];                            \
      Bs[buf][(kc + 2) * 130 + row_] = wreg[rr][2];                            \
      Bs[buf][(kc + 3) * 130 + row_] = wreg[rr][3];                            \
    }                                                                          \
  } while (0)

  // Prologue: stage tile 0
  STAGE_A(0, 0);
  LOAD_W(0);
  STORE_W(0);
  WAIT_ASYNC();
  __syncthreads();

  const int NIT = K / 32;
  for (int it = 0; it < NIT; ++it) {
    const int cur = it & 1;
    const int nxt = cur ^ 1;
    const int knxt = (it + 1) * 32;
    const bool more = knxt < K;

    if (more) {
      STAGE_A(nxt, knxt);  // async copy overlaps with WMMAs below
      LOAD_W(knxt);
    }

    // A fragments: lane holds row m = lane%16, K pairs per ISA layout
    v16bf afr[4];
    #pragma unroll
    for (int mi = 0; mi < 4; mi++) {
      const int rowb = wr * 64 + mi * 16 + ml;
      #pragma unroll
      for (int v = 0; v < 8; v++) {
        const int kk = ((v >> 2) << 4) + (mhalf << 3) + ((v & 3) << 1);
        bf16x2 p = *(const bf16x2*)&As[cur][rowb * 40 + kk];
        afr[mi][2 * v] = p[0]; afr[mi][2 * v + 1] = p[1];
      }
    }
    // B fragments: lane holds column n = lane%16, K pairs per ISA layout
    v16bf bfr[2];
    #pragma unroll
    for (int ni = 0; ni < 2; ni++) {
      const int nb = wc * 32 + ni * 16;
      #pragma unroll
      for (int v = 0; v < 8; v++) {
        const int kk = ((v >> 2) << 4) + (mhalf << 3) + ((v & 3) << 1);
        bfr[ni][2 * v] = Bs[cur][kk * 130 + nb + ml];
        bfr[ni][2 * v + 1] = Bs[cur][(kk + 1) * 130 + nb + ml];
      }
    }
    #pragma unroll
    for (int mi = 0; mi < 4; mi++)
      #pragma unroll
      for (int ni = 0; ni < 2; ni++)
        acc[mi][ni] = wmma_bf16(afr[mi], bfr[ni], acc[mi][ni]);

    if (more) STORE_W(nxt);
    WAIT_ASYNC();
    __syncthreads();
  }

  // Epilogue: C layout lane = n + 16*(m/8), vgpr = m%8
  #pragma unroll
  for (int mi = 0; mi < 4; mi++)
    #pragma unroll
    for (int ni = 0; ni < 2; ni++)
      #pragma unroll
      for (int v = 0; v < 8; v++) {
        const int row = m0 + wr * 64 + mi * 16 + mhalf * 8 + v;
        const int col = n0 + wc * 32 + ni * 16 + ml;
        const float val = acc[mi][ni][v];
        if (OUT_BF16) {
          ((__bf16*)out)[(size_t)row * N + col] = (__bf16)val;
        } else {
          ((float*)out)[(size_t)row * N + col] = val + bias[col];
        }
      }
#undef STAGE_A
#undef WAIT_ASYNC
#undef LOAD_W
#undef STORE_W
}

// ---------------------------------------------------------------------------
// Windowed flash attention. 4 waves/WG; wave handles 16 queries of one (b,h).
// Streams 32-key chunks: S = Q K^T (WMMA), band mask, online softmax, P V.
// ctx written as bf16 (consumed by final WMMA projection).
// ---------------------------------------------------------------------------
__global__ __launch_bounds__(128) void attn_kernel(const __bf16* __restrict__ Q,
                                                   const __bf16* __restrict__ K,
                                                   const __bf16* __restrict__ V,
                                                   __bf16* __restrict__ ctx) {
  __shared__ __bf16 Kt[64 * 34];     // K^T tile: [d][j], stride 34
  __shared__ __bf16 Pw[4][16 * 34];  // per-wave P: [m][key], stride 34

  const int tid  = threadIdx.x;
  const int lane = tid & 31;
  const int wave = tid >> 5;
  const int q0 = blockIdx.x * 64;
  const int h  = blockIdx.y;
  const int b  = blockIdx.z;
  const int qw = q0 + wave * 16;

  const size_t bbase   = (size_t)b * S_ * D_;
  const size_t headoff = (size_t)h * HD_;
  const int mhalf = lane >> 4;
  const int ml = lane & 15;

  // Q A-fragments (16 queries x 64 d -> two 16x32 frags), straight from global
  v16bf qa[2];
  #pragma unroll
  for (int f = 0; f < 2; f++) {
    const __bf16* qrow = Q + bbase + (size_t)(qw + ml) * D_ + headoff + f * 32;
    #pragma unroll
    for (int v = 0; v < 8; v++) {
      const int kk = ((v >> 2) << 4) + (mhalf << 3) + ((v & 3) << 1);
      bf16x2 p = *(const bf16x2*)(qrow + kk);
      qa[f][2 * v] = p[0]; qa[f][2 * v + 1] = p[1];
    }
  }

  v8f o[4];
  float mst[8], lst[8];
  for (int t = 0; t < 4; t++)
    for (int e = 0; e < 8; e++) o[t][e] = 0.0f;
  for (int e = 0; e < 8; e++) { mst[e] = -1e30f; lst[e] = 0.0f; }

  int ks = q0 - (WIN_ - 1);
  if (ks < 0) ks = 0;
  ks &= ~31;
  const int kend = q0 + 63;

  for (int j0 = ks; j0 <= kend; j0 += 32) {
    __syncthreads();
    // Stage K^T: 128 threads transpose a 32(j) x 64(d) tile into Kt[d][j]
    {
      const int j  = tid & 31;
      const int d0 = (tid >> 5) * 16;
      const __bf16* src = K + bbase + (size_t)(j0 + j) * D_ + headoff + d0;
      #pragma unroll
      for (int i = 0; i < 16; i++) Kt[(d0 + i) * 34 + j] = src[i];
    }
    __syncthreads();

    // Scores: two 16x16 tiles (keys j0..+15, j0+16..+31), 2 k-steps over d
    v8f s[2];
    #pragma unroll
    for (int t = 0; t < 2; t++) {
      v8f c;
      for (int e = 0; e < 8; e++) c[e] = 0.0f;
      #pragma unroll
      for (int f = 0; f < 2; f++) {
        v16bf bfrag;
        #pragma unroll
        for (int v = 0; v < 8; v++) {
          const int kk = f * 32 + ((v >> 2) << 4) + (mhalf << 3) + ((v & 3) << 1);
          bfrag[2 * v] = Kt[kk * 34 + t * 16 + ml];
          bfrag[2 * v + 1] = Kt[(kk + 1) * 34 + t * 16 + ml];
        }
        c = wmma_bf16(qa[f], bfrag, c);
      }
      s[t] = c;
    }

    // Band mask + per-row max
    float rmax[8];
    #pragma unroll
    for (int v = 0; v < 8; v++) {
      const int i = qw + mhalf * 8 + v;
      #pragma unroll
      for (int t = 0; t < 2; t++) {
        const int j = j0 + t * 16 + ml;
        const bool ok = (j <= i) && (i - j < WIN_);
        if (!ok) s[t][v] = -1e30f;
      }
      rmax[v] = fmaxf(s[0][v], s[1][v]);
    }
    #pragma unroll
    for (int d = 1; d < 16; d <<= 1)
      #pragma unroll
      for (int v = 0; v < 8; v++)
        rmax[v] = fmaxf(rmax[v], __shfl_xor(rmax[v], d, 32));

    // Online softmax update
    float p0[8], p1[8], rsum[8];
    #pragma unroll
    for (int v = 0; v < 8; v++) {
      const float nm = fmaxf(mst[v], rmax[v]);
      const float sc = __expf(mst[v] - nm);
      p0[v] = __expf(s[0][v] - nm);
      p1[v] = __expf(s[1][v] - nm);
      rsum[v] = p0[v] + p1[v];
      mst[v] = nm;
      lst[v] *= sc;
      #pragma unroll
      for (int t = 0; t < 4; t++) o[t][v] *= sc;
    }
    #pragma unroll
    for (int d = 1; d < 16; d <<= 1)
      #pragma unroll
      for (int v = 0; v < 8; v++) rsum[v] += __shfl_xor(rsum[v], d, 32);
    #pragma unroll
    for (int v = 0; v < 8; v++) lst[v] += rsum[v];

    // P -> LDS (C layout -> row-major), then reload as A fragment
    #pragma unroll
    for (int v = 0; v < 8; v++) {
      const int m = mhalf * 8 + v;
      Pw[wave][m * 34 + ml]      = (__bf16)p0[v];
      Pw[wave][m * 34 + 16 + ml] = (__bf16)p1[v];
    }
    __syncthreads();

    v16bf pa;
    #pragma unroll
    for (int v = 0; v < 8; v++) {
      const int kk = ((v >> 2) << 4) + (mhalf << 3) + ((v & 3) << 1);
      bf16x2 p = *(const bf16x2*)&Pw[wave][ml * 34 + kk];
      pa[2 * v] = p[0]; pa[2 * v + 1] = p[1];
    }
    // V B-fragments straight from global: lane = column n, K pairs per ISA layout
    #pragma unroll
    for (int nt = 0; nt < 4; nt++) {
      v16bf vb;
      #pragma unroll
      for (int v = 0; v < 8; v++) {
        const int kk = ((v >> 2) << 4) + (mhalf << 3) + ((v & 3) << 1);
        const __bf16* vp = V + bbase + (size_t)(j0 + kk) * D_ + headoff + nt * 16 + ml;
        vb[2 * v] = vp[0]; vb[2 * v + 1] = vp[D_];
      }
      o[nt] = wmma_bf16(pa, vb, o[nt]);
    }
  }

  // Normalize and write ctx (bf16)
  #pragma unroll
  for (int v = 0; v < 8; v++) {
    const float inv = 1.0f / lst[v];
    const int row = qw + mhalf * 8 + v;
    #pragma unroll
    for (int nt = 0; nt < 4; nt++)
      ctx[bbase + (size_t)row * D_ + headoff + nt * 16 + ml] =
          (__bf16)(o[nt][v] * inv);
  }
}

// ---------------------------------------------------------------------------
extern "C" void kernel_launch(void* const* d_in, const int* in_sizes, int n_in,
                              void* d_out, int out_size, void* d_ws, size_t ws_size,
                              hipStream_t stream) {
  const float* X  = (const float*)d_in[0];
  const float* Wq = (const float*)d_in[1];
  const float* Wk = (const float*)d_in[2];
  const float* Wv = (const float*)d_in[3];
  const float* Wo = (const float*)d_in[4];
  const float* bo = (const float*)d_in[5];

  const size_t NX = (size_t)B_ * S_ * D_;  // 8,388,608
  const size_t NW = (size_t)D_ * D_;       // 1,048,576

  __bf16* Xb   = (__bf16*)d_ws;
  __bf16* Wqb  = Xb + NX;
  __bf16* Wkb  = Wqb + NW;
  __bf16* Wvb  = Wkb + NW;
  __bf16* Wob  = Wvb + NW;
  __bf16* Qb   = Wob + NW;
  __bf16* Kb   = Qb + NX;
  __bf16* Vb   = Kb + NX;
  __bf16* ctxb = Vb + NX;

  // f32 -> bf16 conversion passes
  cvt_f32_bf16<<<dim3(NX / 1024), 256, 0, stream>>>(X, Xb, (int)NX);
  cvt_f32_bf16<<<dim3(NW / 1024), 256, 0, stream>>>(Wq, Wqb, (int)NW);
  cvt_f32_bf16<<<dim3(NW / 1024), 256, 0, stream>>>(Wk, Wkb, (int)NW);
  cvt_f32_bf16<<<dim3(NW / 1024), 256, 0, stream>>>(Wv, Wvb, (int)NW);
  cvt_f32_bf16<<<dim3(NW / 1024), 256, 0, stream>>>(Wo, Wob, (int)NW);

  const int M = B_ * S_;       // 8192
  dim3 gg(M / 128, D_ / 128);  // (64, 8)

  gemm_xwT<true><<<gg, 256, 0, stream>>>(Xb, Wqb, (void*)Qb, nullptr, M, D_, D_);
  gemm_xwT<true><<<gg, 256, 0, stream>>>(Xb, Wkb, (void*)Kb, nullptr, M, D_, D_);
  gemm_xwT<true><<<gg, 256, 0, stream>>>(Xb, Wvb, (void*)Vb, nullptr, M, D_, D_);

  attn_kernel<<<dim3(S_ / 64, H_, B_), 128, 0, stream>>>(Qb, Kb, Vb, ctxb);

  gemm_xwT<false><<<gg, 256, 0, stream>>>(ctxb, Wob, d_out, bo, M, D_, D_);
}
